// LSTM_9199819948163
// MI455X (gfx1250) — matrix operand order repaired and checked by reading, not compiled, not run on hardware
//
#include <hip/hip_runtime.h>
#include <hip/hip_bf16.h>

// ---------------- problem constants ----------------
#define SEQ    513
#define TSTEPS 512
#define BATCH  128
#define EDIM   512
#define HDIM   1024
#define CDIM   512
#define KTOT   (EDIM + HDIM)   // 1536 : fused [x ; h] K-dimension
#define NG     (4 * HDIM)      // 4096 : g|i|f|o concatenated columns

// ---------------- WMMA vector types ----------------
typedef __attribute__((ext_vector_type(16))) __bf16 v16bf;
typedef __attribute__((ext_vector_type(8)))  __bf16 v8bf;
typedef __attribute__((ext_vector_type(8)))  float  v8f;

union Frag { v16bf v; v8bf h[2]; };

static __device__ __forceinline__ v8f wmma_bf16(const Frag& a, const Frag& b, v8f c) {
    return __builtin_amdgcn_wmma_f32_16x16x32_bf16(
        false, a.v, false, b.v, (short)0, c, false, false);
}

// ---------------- workspace layout (bytes, all 256-aligned) ----------------
#define OFF_EMB    (size_t)0                      // 513*512 bf16      = 525312
#define OFF_WCAT   (size_t)525312                 // 1536*4096 bf16    = 12582912
#define OFF_WP     (size_t)13108224               // 1024*512 bf16     = 1048576
#define OFF_BCAT   (size_t)14156800               // 4096 f32          = 16384
#define OFF_H0     (size_t)14173184               // 128*1024 bf16     = 262144
#define OFF_H1     (size_t)14435328               // 128*1024 bf16     = 262144
#define OFF_C      (size_t)14697472               // 128*1024 f32      = 524288
#define OFF_LOGIT  (size_t)15221760               // 128*512 f32       = 262144

// ---------------- setup kernels ----------------
__global__ void k_cvt_bf16(const float* __restrict__ src, __bf16* __restrict__ dst, int n) {
    int i = blockIdx.x * blockDim.x + threadIdx.x;
    if (i < n) dst[i] = (__bf16)src[i];
}

__global__ void k_build_wcat(const float* __restrict__ Wgx, const float* __restrict__ Wgh,
                             const float* __restrict__ Wix, const float* __restrict__ Wih,
                             const float* __restrict__ Wfx, const float* __restrict__ Wfh,
                             const float* __restrict__ Wox, const float* __restrict__ Woh,
                             __bf16* __restrict__ Wcat) {
    int i = blockIdx.x * blockDim.x + threadIdx.x;
    if (i >= KTOT * NG) return;
    int r = i / NG, cc = i - r * NG;
    int gate = cc >> 10, j = cc & (HDIM - 1);
    const float* Wx[4] = {Wgx, Wix, Wfx, Wox};
    const float* Wh[4] = {Wgh, Wih, Wfh, Woh};
    float v = (r < EDIM) ? Wx[gate][r * HDIM + j] : Wh[gate][(r - EDIM) * HDIM + j];
    Wcat[i] = (__bf16)v;
}

__global__ void k_build_bcat(const float* __restrict__ bg, const float* __restrict__ bi,
                             const float* __restrict__ bf_, const float* __restrict__ bo,
                             float* __restrict__ bcat) {
    int i = blockIdx.x * blockDim.x + threadIdx.x;
    if (i >= NG) return;
    int gate = i >> 10, j = i & (HDIM - 1);
    const float* bs[4] = {bg, bi, bf_, bo};
    bcat[i] = bs[gate][j];
}

__global__ void k_init_state(const float* __restrict__ hinit, const float* __restrict__ cinit,
                             __bf16* __restrict__ h0, float* __restrict__ c) {
    int i = blockIdx.x * blockDim.x + threadIdx.x;
    if (i >= BATCH * HDIM) return;
    int j = i & (HDIM - 1);
    h0[i] = (__bf16)hinit[j];
    c[i]  = cinit[j];
}

// ---------------- one K-slab of the recurrent GEMM ----------------
// 4 A tiles (64 batch rows) x 4 gate B fragments -> 16 WMMAs per slab.
// 8 A-loads + 8 B-loads per 16 WMMAs (1.0 loads/WMMA); each B fragment
// feeds 4 WMMAs, halving L2 weight traffic vs the 2-tile version.
static __device__ __forceinline__ void lstm_k_iter(
    const __bf16* __restrict__ ar0, const __bf16* __restrict__ ar1,
    const __bf16* __restrict__ ar2, const __bf16* __restrict__ ar3,
    const __bf16* __restrict__ Wrow, int klo, v8f acc[16])
{
    Frag b[4];
    #pragma unroll
    for (int g = 0; g < 4; ++g) {
        const __bf16* bp = Wrow + g * HDIM;
        #pragma unroll
        for (int s = 0; s < 8; ++s) {
            b[g].v[s]     = bp[(size_t)(klo + s) * NG];
            b[g].v[8 + s] = bp[(size_t)(klo + 16 + s) * NG];
        }
    }
    Frag a[4];
    a[0].h[0] = *(const v8bf*)(ar0 + klo);
    a[0].h[1] = *(const v8bf*)(ar0 + klo + 16);
    a[1].h[0] = *(const v8bf*)(ar1 + klo);
    a[1].h[1] = *(const v8bf*)(ar1 + klo + 16);
    a[2].h[0] = *(const v8bf*)(ar2 + klo);
    a[2].h[1] = *(const v8bf*)(ar2 + klo + 16);
    a[3].h[0] = *(const v8bf*)(ar3 + klo);
    a[3].h[1] = *(const v8bf*)(ar3 + klo + 16);
    #pragma unroll
    for (int g = 0; g < 4; ++g) {
        #pragma unroll
        for (int mt = 0; mt < 4; ++mt) {
            acc[g * 4 + mt] = wmma_bf16(a[mt], b[g], acc[g * 4 + mt]);
        }
    }
}

// ---------------- one LSTM time-step ----------------
// grid = (64 j-slices of 16 cols, 2 m-groups), block = 32 threads (1 wave).
// Each wave: 4 M-tiles (64 batch rows) x 16 j-cols x 4 gates (16 accumulators).
__global__ void __launch_bounds__(32) k_lstm_step(
    const int* __restrict__ x, const __bf16* __restrict__ emb,
    const __bf16* __restrict__ Wcat, const float* __restrict__ bcat,
    const __bf16* __restrict__ hin, __bf16* __restrict__ hout,
    float* __restrict__ c, int t)
{
    const int lane  = threadIdx.x & 31;
    const int llo   = lane & 15;     // M within tile (A) / j col (B,D)
    const int lhi   = lane >> 4;     // K-half selector
    const int klo   = lhi << 3;      // 0 or 8
    const int kl    = llo + (lhi << 4);  // lane's B row within a 32-K slab
    const int mbase = blockIdx.y << 6;   // 64 rows per wave
    const int jb    = blockIdx.x << 4;   // 16 j-cols per block

    const __bf16* ax[4];
    const __bf16* ah[4];
    #pragma unroll
    for (int mt = 0; mt < 4; ++mt) {
        const int mrow = mbase + (mt << 4) + llo;
        const int xid  = x[mrow * SEQ + t];
        ax[mt] = emb + (size_t)xid * EDIM;
        ah[mt] = hin + (size_t)mrow * HDIM;
    }

    v8f zero = {0.f,0.f,0.f,0.f,0.f,0.f,0.f,0.f};
    v8f acc[16];                     // [gate*4 + mtile]
    #pragma unroll
    for (int i = 0; i < 16; ++i) acc[i] = zero;

    const __bf16* Wbase = Wcat + jb + llo;

    // x-region of the fused K dimension
    for (int kb = 0; kb < EDIM; kb += 32) {
        const __bf16* Wrow = Wbase + (size_t)kb * NG;
        __builtin_prefetch(Wrow + (size_t)32 * NG, 0, 1);
        lstm_k_iter(ax[0] + kb, ax[1] + kb, ax[2] + kb, ax[3] + kb, Wrow, klo, acc);
    }
    // h-region of the fused K dimension
    for (int kb = 0; kb < HDIM; kb += 32) {
        const __bf16* Wrow = Wbase + (size_t)(EDIM + kb) * NG;
        if (kb + 32 < HDIM)
            __builtin_prefetch(Wrow + (size_t)32 * NG, 0, 1);
        lstm_k_iter(ah[0] + kb, ah[1] + kb, ah[2] + kb, ah[3] + kb, Wrow, klo, acc);
    }

    // epilogue: g/i/f/o for element (m, j) are lane/register aligned
    const int j   = jb + llo;
    const float bg  = bcat[j];
    const float bi  = bcat[HDIM + j];
    const float bf_ = bcat[2 * HDIM + j];
    const float bo  = bcat[3 * HDIM + j];
    #pragma unroll
    for (int mt = 0; mt < 4; ++mt) {
        #pragma unroll
        for (int r = 0; r < 8; ++r) {
            const int m = mbase + (mt << 4) + r + (lhi << 3);
            const float gp = acc[0  + mt][r] + bg;
            const float ip = acc[4  + mt][r] + bi;
            const float fp = acc[8  + mt][r] + bf_;
            const float op = acc[12 + mt][r] + bo;
            const float gg = tanhf(gp);
            const float ii = 1.f / (1.f + __expf(-ip));
            const float ff = 1.f / (1.f + __expf(-fp));
            const float oo = 1.f / (1.f + __expf(-op));
            const size_t idx = (size_t)m * HDIM + j;
            const float cn = gg * ii + c[idx] * ff;
            const float hn = tanhf(cn) * oo;
            c[idx]    = cn;
            hout[idx] = (__bf16)hn;
        }
    }
}

// ---------------- final projection: logits = h_T @ Wp + bp ----------------
// grid = (32 n-slices of 16 cols, 2 m-groups), block = 64 threads (2 waves).
__global__ void __launch_bounds__(64) k_proj(
    const __bf16* __restrict__ hfin, const __bf16* __restrict__ Wp,
    const float* __restrict__ bp, float* __restrict__ logits)
{
    const int tid   = threadIdx.x;
    const int w     = tid >> 5;
    const int lane  = tid & 31;
    const int llo   = lane & 15;
    const int lhi   = lane >> 4;
    const int klo   = lhi << 3;
    const int kl    = llo + (lhi << 4);
    const int mbase = (blockIdx.y << 6) + (w << 5);
    const int nb    = blockIdx.x << 4;

    const __bf16* a0r = hfin + (size_t)(mbase + llo) * HDIM;
    const __bf16* a1r = hfin + (size_t)(mbase + 16 + llo) * HDIM;

    v8f zero = {0.f,0.f,0.f,0.f,0.f,0.f,0.f,0.f};
    v8f acc[2] = {zero, zero};

    for (int kb = 0; kb < HDIM; kb += 32) {
        Frag b;
        const __bf16* bpp = Wp + (size_t)kb * CDIM + nb + llo;
        #pragma unroll
        for (int s = 0; s < 8; ++s) {
            b.v[s]     = bpp[(size_t)(klo + s) * CDIM];
            b.v[8 + s] = bpp[(size_t)(klo + 16 + s) * CDIM];
        }
        Frag a0, a1;
        a0.h[0] = *(const v8bf*)(a0r + kb + klo);
        a0.h[1] = *(const v8bf*)(a0r + kb + klo + 16);
        a1.h[0] = *(const v8bf*)(a1r + kb + klo);
        a1.h[1] = *(const v8bf*)(a1r + kb + klo + 16);
        acc[0] = wmma_bf16(a0, b, acc[0]);
        acc[1] = wmma_bf16(a1, b, acc[1]);
    }

    const int n = nb + llo;
    const float bias = bp[n];
    #pragma unroll
    for (int mt = 0; mt < 2; ++mt) {
        #pragma unroll
        for (int r = 0; r < 8; ++r) {
            const int m = mbase + (mt << 4) + r + (lhi << 3);
            logits[(size_t)m * CDIM + n] = acc[mt][r] + bias;
        }
    }
}

// ---------------- log_softmax over C=512 per batch row ----------------
__global__ void __launch_bounds__(256) k_logsoftmax(const float* __restrict__ logits,
                                                    float* __restrict__ out) {
    const int row = blockIdx.x;
    const int tid = threadIdx.x;
    __shared__ float red[256];

    const float v0 = logits[(size_t)row * CDIM + tid];
    const float v1 = logits[(size_t)row * CDIM + tid + 256];

    red[tid] = fmaxf(v0, v1);
    __syncthreads();
    for (int s = 128; s > 0; s >>= 1) {
        if (tid < s) red[tid] = fmaxf(red[tid], red[tid + s]);
        __syncthreads();
    }
    const float mx = red[0];
    __syncthreads();

    red[tid] = __expf(v0 - mx) + __expf(v1 - mx);
    __syncthreads();
    for (int s = 128; s > 0; s >>= 1) {
        if (tid < s) red[tid] += red[tid + s];
        __syncthreads();
    }
    const float lse = mx + logf(red[0]);

    out[(size_t)row * CDIM + tid]       = v0 - lse;
    out[(size_t)row * CDIM + tid + 256] = v1 - lse;
}

// ---------------- host-side launch ----------------
extern "C" void kernel_launch(void* const* d_in, const int* in_sizes, int n_in,
                              void* d_out, int out_size, void* d_ws, size_t ws_size,
                              hipStream_t stream) {
    (void)in_sizes; (void)n_in; (void)out_size; (void)ws_size;
    const int*   d_x    = (const int*)  d_in[0];
    const float* d_emb  = (const float*)d_in[1];
    const float* W_gx   = (const float*)d_in[2];
    const float* W_gh   = (const float*)d_in[3];
    const float* b_g    = (const float*)d_in[4];
    const float* W_ix   = (const float*)d_in[5];
    const float* W_ih   = (const float*)d_in[6];
    const float* b_i    = (const float*)d_in[7];
    const float* W_fx   = (const float*)d_in[8];
    const float* W_fh   = (const float*)d_in[9];
    const float* b_f    = (const float*)d_in[10];
    const float* W_ox   = (const float*)d_in[11];
    const float* W_oh   = (const float*)d_in[12];
    const float* b_o    = (const float*)d_in[13];
    const float* W_ph   = (const float*)d_in[14];
    const float* b_p    = (const float*)d_in[15];
    const float* h_init = (const float*)d_in[16];
    const float* c_init = (const float*)d_in[17];

    char* ws = (char*)d_ws;
    __bf16* embb  = (__bf16*)(ws + OFF_EMB);
    __bf16* Wcat  = (__bf16*)(ws + OFF_WCAT);
    __bf16* Wpb   = (__bf16*)(ws + OFF_WP);
    float*  bcat  = (float*) (ws + OFF_BCAT);
    __bf16* h0    = (__bf16*)(ws + OFF_H0);
    __bf16* h1    = (__bf16*)(ws + OFF_H1);
    float*  cbuf  = (float*) (ws + OFF_C);
    float*  logit = (float*) (ws + OFF_LOGIT);

    // setup: bf16 conversions, fused weight/bias tensors, initial state
    k_cvt_bf16  <<<(SEQ * EDIM + 255) / 256, 256, 0, stream>>>(d_emb, embb, SEQ * EDIM);
    k_build_wcat<<<(KTOT * NG  + 255) / 256, 256, 0, stream>>>(W_gx, W_gh, W_ix, W_ih,
                                                               W_fx, W_fh, W_ox, W_oh, Wcat);
    k_cvt_bf16  <<<(HDIM * CDIM + 255) / 256, 256, 0, stream>>>(W_ph, Wpb, HDIM * CDIM);
    k_build_bcat<<<(NG + 255) / 256, 256, 0, stream>>>(b_g, b_i, b_f, b_o, bcat);
    k_init_state<<<(BATCH * HDIM + 255) / 256, 256, 0, stream>>>(h_init, c_init, h0, cbuf);

    // 512 sequential recurrent steps, h double-buffered (t even: h0->h1, t odd: h1->h0)
    dim3 sgrid(HDIM / 16, BATCH / 64);
    for (int t = 0; t < TSTEPS; ++t) {
        __bf16* hin  = (t & 1) ? h1 : h0;
        __bf16* hout = (t & 1) ? h0 : h1;
        k_lstm_step<<<sgrid, 32, 0, stream>>>(d_x, embb, Wcat, bcat, hin, hout, cbuf, t);
    }

    // t=511 wrote into h0 -> final hidden state
    dim3 pgrid(CDIM / 16, 2);
    k_proj      <<<pgrid, 64, 0, stream>>>(h0, Wpb, b_p, logit);
    k_logsoftmax<<<BATCH, 256, 0, stream>>>(logit, (float*)d_out);
}
